// RealizedGARCHPINNv2_1254130451033
// MI455X (gfx1250) — matrix operand order repaired and checked by reading, not compiled, not run on hardware
//
#include <hip/hip_runtime.h>
#include <hip/hip_bf16.h>
#include <math.h>

// ---------------------------------------------------------------------------
// RealizedGARCH-PINN forward scan, MI455X (gfx1250) persistent single-WGP
// design. B=256, T=1024, H=128, IN=5.
//
// One workgroup of 1024 threads (32 wave32 waves) stays resident for all
// T steps; all weights/activations live in LDS. The 256x128x128 GEMM per
// step runs on v_wmma_f32_16x16x32_bf16. Operand arrays use a padded row
// stride (136 bf16 = 68 dwords) so per-lane row-strided ds_load_b128
// fragment loads tile all 64 LDS banks instead of hammering one group.
// ---------------------------------------------------------------------------

typedef __attribute__((ext_vector_type(16))) __bf16 v16bf;
typedef __attribute__((ext_vector_type(8)))  __bf16 v8bf;
typedef __attribute__((ext_vector_type(8)))  float  v8f;

constexpr int Bn = 256, Tn = 1024, Hn = 128, INn = 5;
constexpr int AP = 136;                   // padded row stride (bf16 elems), 272 B

// ---- LDS carve (bytes) ----
constexpr int OFF_W2BF  = 0;              // 128*136*2 = 34816
constexpr int OFF_ABF   = 34816;          // 256*136*2 = 69632 -> 104448
constexpr int OFF_W1    = 104448;         // 128*5 f32 = 2560
constexpr int OFF_B1    = 107008;         // 128 f32 each below
constexpr int OFF_G1    = 107520;
constexpr int OFF_BE1   = 108032;
constexpr int OFF_B2    = 108544;
constexpr int OFF_G2    = 109056;
constexpr int OFF_BE2   = 109568;
constexpr int OFF_W3    = 110080;
constexpr int OFF_FEATS = 110592;         // 256*5 f32 = 5120
constexpr int OFF_CLH   = 115712;         // 256 f32 carry enh
constexpr int OFF_CZ    = 116736;         // 256 f32 carry z
constexpr int OFF_CLX   = 117760;         // 256 f32 carry log_x
constexpr int OFF_LOGH  = 118784;         // 256 f32 log_h_t
constexpr int OFF_S1    = 119808;         // 128 f32 sum   (BN1)
constexpr int OFF_Q1    = 120320;         // 128 f32 sumsq (BN1)
constexpr int OFF_S2    = 120832;         // 128 f32 sum   (BN2)
constexpr int OFF_Q2    = 121344;         // 128 f32 sumsq (BN2)
constexpr int OFF_RS    = 121856;         // 256 f32 rowsum (GEMM3)
constexpr int OFF_PARS  = 122880;         // scalar params
constexpr int SMEM_BYTES = 122944;

struct GPtrs {
  const float *ret, *lrv;
  const float *W1, *b1, *g1, *be1;
  const float *W2, *b2, *g2, *be2;
  const float *W3, *b3;
  const float *omega, *beta, *tau1, *tau2, *gamma, *xi, *phi, *d1, *d2, *mu;
  float *o_lh, *o_lx, *o_z, *o_u;
};

__device__ __forceinline__ float clampf(float x, float lo, float hi) {
  return fminf(fmaxf(x, lo), hi);
}
__device__ __forceinline__ float lrelu(float x) { return x > 0.0f ? x : 0.1f * x; }

__device__ __forceinline__ v16bf ld_frag(const __bf16* base0, const __bf16* base1) {
  v8bf lo = *(const v8bf*)base0;
  v8bf hi = *(const v8bf*)base1;
  v16bf f;
  #pragma unroll
  for (int i = 0; i < 8; i++) { f[i] = lo[i]; f[i + 8] = hi[i]; }
  return f;
}

__global__ __launch_bounds__(1024)
void garch_pinn_kernel(GPtrs P) {
  extern __shared__ __align__(16) char smem[];
  __bf16* W2bf = (__bf16*)(smem + OFF_W2BF);
  __bf16* Abf  = (__bf16*)(smem + OFF_ABF);
  float* W1s  = (float*)(smem + OFF_W1);
  float* b1s  = (float*)(smem + OFF_B1);
  float* g1s  = (float*)(smem + OFF_G1);
  float* be1s = (float*)(smem + OFF_BE1);
  float* b2s  = (float*)(smem + OFF_B2);
  float* g2s  = (float*)(smem + OFF_G2);
  float* be2s = (float*)(smem + OFF_BE2);
  float* W3s  = (float*)(smem + OFF_W3);
  float* feats   = (float*)(smem + OFF_FEATS);
  float* carryLH = (float*)(smem + OFF_CLH);
  float* carryZ  = (float*)(smem + OFF_CZ);
  float* carryLX = (float*)(smem + OFF_CLX);
  float* logh    = (float*)(smem + OFF_LOGH);
  float* s1 = (float*)(smem + OFF_S1);
  float* q1 = (float*)(smem + OFF_Q1);
  float* s2 = (float*)(smem + OFF_S2);
  float* q2 = (float*)(smem + OFF_Q2);
  float* rowsum = (float*)(smem + OFF_RS);
  float* pars   = (float*)(smem + OFF_PARS);

  const int tid  = threadIdx.x;
  const int lane = tid & 31;
  const int wid  = tid >> 5;            // 0..31
  const int mt   = wid >> 1;            // 0..15   (M tile of this wave)
  const int ntb  = (wid & 1) * 4;       // 0 or 4  (first N tile of this wave)
  const int nloc = lane & 15;
  const int gh   = lane >> 4;           // lane half (ISA layout group)

  // ---------------- preload weights into LDS ----------------
  for (int i = tid; i < Hn * Hn; i += 1024) {
    int n = i >> 7, k = i & 127;
    W2bf[n * AP + k] = (__bf16)P.W2[i];  // W2 row-major [n][k]; row n == B column n
  }
  for (int i = tid; i < Hn * INn; i += 1024) W1s[i] = P.W1[i];
  if (tid < Hn) {
    b1s[tid] = P.b1[tid];  g1s[tid] = P.g1[tid];  be1s[tid] = P.be1[tid];
    b2s[tid] = P.b2[tid];  g2s[tid] = P.g2[tid];  be2s[tid] = P.be2[tid];
    W3s[tid] = P.W3[tid];
  }
  if (tid == 0) {
    pars[0] = P.omega[0]; pars[1] = P.beta[0]; pars[2] = P.tau1[0];
    pars[3] = P.tau2[0];  pars[4] = P.gamma[0]; pars[5] = P.xi[0];
    pars[6] = P.phi[0];   pars[7] = P.d1[0];   pars[8] = P.d2[0];
    pars[9] = P.mu[0];    pars[10] = P.b3[0];
  }
  __syncthreads();

  // per-thread GEMM1 weights (thread handles feature f1, row group rg)
  const int f1 = tid & 127, rg = tid >> 7;
  float w1r[5];
  #pragma unroll
  for (int i = 0; i < 5; i++) w1r[i] = W1s[f1 * 5 + i];
  const float b1r = b1s[f1];

  // fragment base pointers (constant across t)
  const __bf16* aRow = Abf + (mt * 16 + nloc) * AP;          // + kt*32 + 16*gh / +8
  const __bf16* bRow0 = W2bf + ((ntb + 0) * 16 + nloc) * AP + 16 * gh;
  const __bf16* bRow1 = W2bf + ((ntb + 1) * 16 + nloc) * AP + 16 * gh;
  const __bf16* bRow2 = W2bf + ((ntb + 2) * 16 + nloc) * AP + 16 * gh;
  const __bf16* bRow3 = W2bf + ((ntb + 3) * 16 + nloc) * AP + 16 * gh;

  // ---------------- sequential scan ----------------
  for (int t = 0; t < Tn; t++) {
    const float tod = (float)(t % 78) * (1.0f / 77.0f);
    const float dow = (float)((t / 78) % 5) * 0.25f;

    if (tid < Hn) { s1[tid] = 0.f; q1[tid] = 0.f; s2[tid] = 0.f; q2[tid] = 0.f; }
    if (tid < Bn) {
      rowsum[tid] = 0.f;
      // ---- GARCH recursion ----
      float lh_t;
      if (t == 0) {
        lh_t = clampf(P.lrv[tid * Tn], -10.f, 2.f);  // log(clip(exp(clip(.)),1e-6,100))
      } else {
        float zc  = clampf(carryZ[tid], -10.f, 10.f);
        float z2  = fminf(zc * zc, 100.f);
        float tau = pars[2] * zc + pars[3] * (z2 - 1.f);
        lh_t = clampf(pars[0] + pars[1] * clampf(carryLH[tid], -10.f, 2.f) + tau
                      + pars[4] * clampf(carryLX[tid], -10.f, 2.f), -10.f, 2.f);
      }
      float h   = expf(lh_t);
      float hcl = clampf(h, 1e-6f, 100.f);
      float r   = P.ret[tid * Tn + t];
      float z   = clampf((r - pars[9]) / (sqrtf(hcl) + 1e-8f), -10.f, 10.f);
      float z2b = fminf(z * z, 100.f);
      float lx  = clampf(pars[5] + pars[6] * lh_t + pars[7] * z + pars[8] * (z2b - 1.f),
                         -10.f, 2.f);
      float u   = clampf(P.lrv[tid * Tn + t], -10.f, 2.f) - lx;
      logh[tid] = lh_t; carryZ[tid] = z; carryLX[tid] = lx;
      float* ft = &feats[tid * 5];
      ft[0] = lh_t; ft[1] = z; ft[2] = u; ft[3] = tod; ft[4] = dow;
      P.o_lx[tid * Tn + t] = lx;
      P.o_z [tid * Tn + t] = z;
      P.o_u [tid * Tn + t] = u;
    }
    __syncthreads();

    // ---- GEMM1 (256x5 @ 5x128) + leaky + batch stats ----
    float act[32];
    {
      float ls = 0.f, lq = 0.f;
      #pragma unroll
      for (int r = 0; r < 32; r++) {
        const float* ft = &feats[(rg * 32 + r) * 5];
        float v = b1r;
        #pragma unroll
        for (int i = 0; i < 5; i++) v = fmaf(ft[i], w1r[i], v);
        v = lrelu(v);
        act[r] = v; ls += v; lq += v * v;
      }
      atomicAdd(&s1[f1], ls);
      atomicAdd(&q1[f1], lq);
    }
    __syncthreads();
    {
      float mean = s1[f1] * (1.f / 256.f);
      float var  = q1[f1] * (1.f / 256.f) - mean * mean;
      float rs   = rsqrtf(var + 1e-5f);
      float sc   = rs * g1s[f1];
      float sh   = be1s[f1] - mean * sc;
      #pragma unroll
      for (int r = 0; r < 32; r++)
        Abf[(rg * 32 + r) * AP + f1] = (__bf16)(act[r] * sc + sh);
    }
    __syncthreads();

    // ---- GEMM2 (256x128 @ 128x128) via v_wmma_f32_16x16x32_bf16 ----
    v8f acc[4];
    #pragma unroll
    for (int i = 0; i < 4; i++)
      #pragma unroll
      for (int j = 0; j < 8; j++) acc[i][j] = 0.0f;

    #pragma unroll
    for (int kt = 0; kt < 4; kt++) {
      // issue ALL fragment loads for this K-step first, then the 4 WMMAs,
      // so the scheduler can use partial s_wait_dscnt values.
      const __bf16* ar = aRow + kt * 32;
      v16bf afr = ld_frag(ar + 16 * gh, ar + 16 * gh + 8);
      v16bf bf0 = ld_frag(bRow0 + kt * 32, bRow0 + kt * 32 + 8);
      v16bf bf1 = ld_frag(bRow1 + kt * 32, bRow1 + kt * 32 + 8);
      v16bf bf2 = ld_frag(bRow2 + kt * 32, bRow2 + kt * 32 + 8);
      v16bf bf3 = ld_frag(bRow3 + kt * 32, bRow3 + kt * 32 + 8);
      acc[0] = __builtin_amdgcn_wmma_f32_16x16x32_bf16(false, afr, false, bf0,
                                                       (short)0, acc[0], false, false);
      acc[1] = __builtin_amdgcn_wmma_f32_16x16x32_bf16(false, afr, false, bf1,
                                                       (short)0, acc[1], false, false);
      acc[2] = __builtin_amdgcn_wmma_f32_16x16x32_bf16(false, afr, false, bf2,
                                                       (short)0, acc[2], false, false);
      acc[3] = __builtin_amdgcn_wmma_f32_16x16x32_bf16(false, afr, false, bf3,
                                                       (short)0, acc[3], false, false);
    }

    // ---- bias + leaky + BN2 stats ----
    #pragma unroll
    for (int i = 0; i < 4; i++) {
      int ng = (ntb + i) * 16 + nloc;
      float bb = b2s[ng];
      float s = 0.f, q = 0.f;
      #pragma unroll
      for (int j = 0; j < 8; j++) {
        float v = lrelu(acc[i][j] + bb);
        acc[i][j] = v; s += v; q += v * v;
      }
      atomicAdd(&s2[ng], s);
      atomicAdd(&q2[ng], q);
    }
    __syncthreads();

    // ---- BN2 normalize + GEMM3 partial dot with W3 ----
    float p[8];
    #pragma unroll
    for (int j = 0; j < 8; j++) p[j] = 0.f;
    #pragma unroll
    for (int i = 0; i < 4; i++) {
      int ng = (ntb + i) * 16 + nloc;
      float mean = s2[ng] * (1.f / 256.f);
      float var  = q2[ng] * (1.f / 256.f) - mean * mean;
      float rs   = rsqrtf(var + 1e-5f);
      float sc   = rs * g2s[ng];
      float sh   = be2s[ng] - mean * sc;
      float w3   = W3s[ng];
      #pragma unroll
      for (int j = 0; j < 8; j++) p[j] += (acc[i][j] * sc + sh) * w3;
    }
    // reduce over the 16 columns held in each lane half
    #pragma unroll
    for (int off = 1; off < 16; off <<= 1)
      #pragma unroll
      for (int j = 0; j < 8; j++) p[j] += __shfl_xor(p[j], off, 32);
    if (nloc == 0) {
      #pragma unroll
      for (int j = 0; j < 8; j++)
        atomicAdd(&rowsum[mt * 16 + 8 * gh + j], p[j]);
    }
    __syncthreads();

    // ---- tanh head, enhance, write carry + output ----
    if (tid < Bn) {
      float o   = tanhf(rowsum[tid] + pars[10]);
      float enh = clampf(logh[tid] + 0.005f * o, -10.f, 2.f);
      P.o_lh[tid * Tn + t] = enh;
      carryLH[tid] = enh;
    }
    __syncthreads();
  }
}

extern "C" void kernel_launch(void* const* d_in, const int* in_sizes, int n_in,
                              void* d_out, int out_size, void* d_ws, size_t ws_size,
                              hipStream_t stream) {
  (void)in_sizes; (void)n_in; (void)out_size; (void)d_ws; (void)ws_size;
  GPtrs P;
  P.ret = (const float*)d_in[0];   P.lrv = (const float*)d_in[1];
  P.W1  = (const float*)d_in[2];   P.b1  = (const float*)d_in[3];
  P.g1  = (const float*)d_in[4];   P.be1 = (const float*)d_in[5];
  P.W2  = (const float*)d_in[6];   P.b2  = (const float*)d_in[7];
  P.g2  = (const float*)d_in[8];   P.be2 = (const float*)d_in[9];
  P.W3  = (const float*)d_in[10];  P.b3  = (const float*)d_in[11];
  P.omega = (const float*)d_in[12]; P.beta = (const float*)d_in[13];
  P.tau1  = (const float*)d_in[14]; P.tau2 = (const float*)d_in[15];
  P.gamma = (const float*)d_in[16]; P.xi   = (const float*)d_in[17];
  P.phi   = (const float*)d_in[18]; P.d1   = (const float*)d_in[19];
  P.d2    = (const float*)d_in[20]; P.mu   = (const float*)d_in[21];
  float* out = (float*)d_out;
  P.o_lh = out;
  P.o_lx = out + (size_t)Bn * Tn;
  P.o_z  = out + (size_t)2 * Bn * Tn;
  P.o_u  = out + (size_t)3 * Bn * Tn;
  hipLaunchKernelGGL(garch_pinn_kernel, dim3(1), dim3(1024), SMEM_BYTES, stream, P);
}
